// PlacementOnlyNetwork_22574348107936
// MI455X (gfx1250) — compile-verified, matrix operand order checked by reading, not run on hardware
//
#include <hip/hip_runtime.h>
#include <hip/hip_bf16.h>

// Model dims
constexpr int B_  = 32;
constexpr int S_  = 256;
constexpr int H_  = 256;
constexpr int DIN_= 16;
constexpr int NH_ = 8;
constexpr int NL_ = 3;
constexpr int M_  = B_ * S_;   // 8192 rows for the big GEMMs

typedef __attribute__((ext_vector_type(16))) _Float16 v16h;
typedef __attribute__((ext_vector_type(4)))  _Float16 v4h;
typedef __attribute__((ext_vector_type(8)))  float    v8f;

// ---------------------------------------------------------------------------
// WMMA helpers (CDNA5 v_wmma_f32_16x16x32_f16, wave32)
// A-frag: lane l holds row (l&15); element e<8 -> K = 8*(l>>4)+e,
//         e>=8 -> K = 16 + 8*(l>>4) + (e-8).
// B-frag for A@W^T: lane l holds row (l&15) of W with identical K indexing.
// C/D:    element j -> row j + 8*(l>>4), col (l&15).
// ---------------------------------------------------------------------------
__device__ __forceinline__ v16h frag_h(const _Float16* row, int hsel) {
  v16h f;
  const _Float16* p0 = row + 8 * hsel;
  const _Float16* p1 = row + 16 + 8 * hsel;
#pragma unroll
  for (int e = 0; e < 8; ++e) { f[e] = p0[e]; f[8 + e] = p1[e]; }
  return f;
}
__device__ __forceinline__ v16h frag_f(const float* row, int hsel) {
  v16h f;
  const float* p0 = row + 8 * hsel;
  const float* p1 = row + 16 + 8 * hsel;
#pragma unroll
  for (int e = 0; e < 8; ++e) { f[e] = (_Float16)p0[e]; f[8 + e] = (_Float16)p1[e]; }
  return f;
}
__device__ __forceinline__ v8f wmma32(v16h a, v16h b, v8f c) {
  return __builtin_amdgcn_wmma_f32_16x16x32_f16(false, a, false, b, (short)0, c,
                                                false, false);
}
__device__ __forceinline__ v4h cvt4(float4 v) {
  v4h h;
  h[0] = (_Float16)v.x; h[1] = (_Float16)v.y;
  h[2] = (_Float16)v.z; h[3] = (_Float16)v.w;
  return h;
}

// ---------------------------------------------------------------------------
// Generic fused GEMM: C[M,N] = act(A[M,K] @ W[N,K]^T + bias + res)
// Block tile 64x64, 128 threads (4 waves), each wave a 32x32 sub-tile.
// PADK=false: K % 32 == 0; staging fully unrolled: 8 outstanding
//             global_load_b128 before any wait, then cvt + ds_store_b64.
// PADK=true:  scalar zero-padded staging (the single K=16 GEMM).
// ---------------------------------------------------------------------------
template <bool PADK>
__global__ __launch_bounds__(128) void gemm_wmma_kernel(
    const float* __restrict__ A, int lda,
    const float* __restrict__ W, int ldw,
    const float* __restrict__ bias,
    const float* __restrict__ res, int ldr,
    float* __restrict__ C, int ldc,
    int K, int act)
{
  __shared__ _Float16 sA[64][40];
  __shared__ _Float16 sW[64][40];
  const int tid = threadIdx.x;
  const int lane = tid & 31, wv = tid >> 5;
  const int bm0 = blockIdx.y * 64, bn0 = blockIdx.x * 64;
  const int m0w = (wv & 1) * 32, n0w = (wv >> 1) * 32;
  v8f acc[2][2] = {};

  for (int kt = 0; kt < K; kt += 32) {
    if (PADK) {
      for (int i = tid; i < 64 * 32; i += 128) {
        int r = i >> 5, cc = i & 31;
        int k = kt + cc;
        sA[r][cc] = (_Float16)((k < K) ? A[(size_t)(bm0 + r) * lda + k] : 0.f);
        sW[r][cc] = (_Float16)((k < K) ? W[(size_t)(bn0 + r) * ldw + k] : 0.f);
      }
    } else {
      // 64 rows x 8 float4 per tile. Batch: issue all 8 loads, then store.
      float4 ar[4], wr[4];
#pragma unroll
      for (int u = 0; u < 4; ++u) {
        int i = tid + u * 128;
        int r = i >> 3, c4 = (i & 7) << 2;
        ar[u] = *(const float4*)&A[(size_t)(bm0 + r) * lda + kt + c4];
        wr[u] = *(const float4*)&W[(size_t)(bn0 + r) * ldw + kt + c4];
      }
#pragma unroll
      for (int u = 0; u < 4; ++u) {
        int i = tid + u * 128;
        int r = i >> 3, c4 = (i & 7) << 2;
        *(v4h*)&sA[r][c4] = cvt4(ar[u]);
        *(v4h*)&sW[r][c4] = cvt4(wr[u]);
      }
    }
    if (kt + 32 < K) {  // warm L2 for the next K-slab (global_prefetch_b8)
      __builtin_prefetch(&A[(size_t)(bm0 + (tid >> 1)) * lda + kt + 32], 0, 1);
      __builtin_prefetch(&W[(size_t)(bn0 + (tid >> 1)) * ldw + kt + 32], 0, 1);
    }
    __syncthreads();
    v16h a0 = frag_h(&sA[m0w + (lane & 15)][0], lane >> 4);
    v16h a1 = frag_h(&sA[m0w + 16 + (lane & 15)][0], lane >> 4);
    v16h b0 = frag_h(&sW[n0w + (lane & 15)][0], lane >> 4);
    v16h b1 = frag_h(&sW[n0w + 16 + (lane & 15)][0], lane >> 4);
    acc[0][0] = wmma32(a0, b0, acc[0][0]);
    acc[0][1] = wmma32(a0, b1, acc[0][1]);
    acc[1][0] = wmma32(a1, b0, acc[1][0]);
    acc[1][1] = wmma32(a1, b1, acc[1][1]);
    __syncthreads();
  }
#pragma unroll
  for (int i = 0; i < 2; ++i)
#pragma unroll
    for (int j = 0; j < 2; ++j) {
      int n  = bn0 + n0w + j * 16 + (lane & 15);
      int mb = bm0 + m0w + i * 16 + ((lane >> 4) << 3);
      float bv = bias ? bias[n] : 0.f;
#pragma unroll
      for (int r = 0; r < 8; ++r) {
        float v = acc[i][j][r] + bv;
        if (res) v += res[(size_t)(mb + r) * ldr + n];
        if (act) v = fmaxf(v, 0.f);
        C[(size_t)(mb + r) * ldc + n] = v;
      }
    }
}

// ---------------------------------------------------------------------------
// Small-M GEMM (M == 32) for the per-step LSTM/ptr projections.
// C[32,N] = A[32,K] @ W[N,K]^T + bias + pre  (pre has row stride ldp)
// Block tile 32x64, 128 threads (4 waves), 1x2 frags per wave. K % 32 == 0.
// ---------------------------------------------------------------------------
__global__ __launch_bounds__(128) void gemm_small_kernel(
    const float* __restrict__ A, int lda,
    const float* __restrict__ W, int ldw,
    const float* __restrict__ bias,
    const float* __restrict__ pre, int ldp,
    float* __restrict__ C, int ldc,
    int K)
{
  __shared__ _Float16 sA[32][40];
  __shared__ _Float16 sW[64][40];
  const int tid = threadIdx.x;
  const int lane = tid & 31, wv = tid >> 5;
  const int bn0 = blockIdx.x * 64;
  const int m0w = (wv & 1) * 16, n0w = (wv >> 1) * 32;
  v8f acc[2] = {};

  for (int kt = 0; kt < K; kt += 32) {
    // A: 32x8 = 256 vec4s (2/thread); W: 64x8 = 512 vec4s (4/thread).
    float4 ar[2], wr[4];
#pragma unroll
    for (int u = 0; u < 2; ++u) {
      int i = tid + u * 128;
      int r = i >> 3, c4 = (i & 7) << 2;
      ar[u] = *(const float4*)&A[(size_t)r * lda + kt + c4];
    }
#pragma unroll
    for (int u = 0; u < 4; ++u) {
      int i = tid + u * 128;
      int r = i >> 3, c4 = (i & 7) << 2;
      wr[u] = *(const float4*)&W[(size_t)(bn0 + r) * ldw + kt + c4];
    }
#pragma unroll
    for (int u = 0; u < 2; ++u) {
      int i = tid + u * 128;
      int r = i >> 3, c4 = (i & 7) << 2;
      *(v4h*)&sA[r][c4] = cvt4(ar[u]);
    }
#pragma unroll
    for (int u = 0; u < 4; ++u) {
      int i = tid + u * 128;
      int r = i >> 3, c4 = (i & 7) << 2;
      *(v4h*)&sW[r][c4] = cvt4(wr[u]);
    }
    __syncthreads();
    v16h a  = frag_h(&sA[m0w + (lane & 15)][0], lane >> 4);
    v16h b0 = frag_h(&sW[n0w + (lane & 15)][0], lane >> 4);
    v16h b1 = frag_h(&sW[n0w + 16 + (lane & 15)][0], lane >> 4);
    acc[0] = wmma32(a, b0, acc[0]);
    acc[1] = wmma32(a, b1, acc[1]);
    __syncthreads();
  }
#pragma unroll
  for (int j = 0; j < 2; ++j) {
    int n  = bn0 + n0w + j * 16 + (lane & 15);
    int mb = m0w + ((lane >> 4) << 3);
    float bv = bias ? bias[n] : 0.f;
#pragma unroll
    for (int r = 0; r < 8; ++r) {
      float v = acc[j][r] + bv;
      if (pre) v += pre[(size_t)(mb + r) * ldp + n];
      C[(size_t)(mb + r) * ldc + n] = v;
    }
  }
}

// ---------------------------------------------------------------------------
// Attention: one block per (b, head, 32-query tile). WMMA QK^T -> softmax ->
// WMMA PV. V tile staged transposed, reusing the K-tile LDS slot.
// ---------------------------------------------------------------------------
__global__ __launch_bounds__(128) void attn_kernel(
    const float* __restrict__ qkv, float* __restrict__ out)
{
  __shared__ __align__(16) char smem[2560 + 33792 + 20480];
  _Float16 (*sQ)[40]   = (_Float16(*)[40])(smem);
  float    (*sS)[264]  = (float(*)[264])(smem + 2560);
  _Float16 (*sK)[40]   = (_Float16(*)[40])(smem + 2560 + 33792);
  _Float16 (*sVt)[264] = (_Float16(*)[264])(smem + 2560 + 33792);  // reuses sK

  const int tid = threadIdx.x, lane = tid & 31, wv = tid >> 5;
  const int b = blockIdx.z, hh = blockIdx.y, q0 = blockIdx.x * 32;

  {  // Q tile: 32x8 vec4s (2/thread), batched
    float4 qr[2];
#pragma unroll
    for (int u = 0; u < 2; ++u) {
      int i = tid + u * 128;
      int r = i >> 3, c4 = (i & 7) << 2;
      qr[u] = *(const float4*)&qkv[(size_t)(b * S_ + q0 + r) * 768 + hh * 32 + c4];
    }
#pragma unroll
    for (int u = 0; u < 2; ++u) {
      int i = tid + u * 128;
      int r = i >> 3, c4 = (i & 7) << 2;
      *(v4h*)&sQ[r][c4] = cvt4(qr[u]);
    }
  }
  // K tile: 256x8 vec4s (16/thread), batched in groups of 4
#pragma unroll
  for (int g = 0; g < 16; g += 4) {
    float4 kr[4];
#pragma unroll
    for (int u = 0; u < 4; ++u) {
      int i = tid + (g + u) * 128;
      int r = i >> 3, c4 = (i & 7) << 2;
      kr[u] = *(const float4*)&qkv[(size_t)(b * S_ + r) * 768 + 256 + hh * 32 + c4];
    }
#pragma unroll
    for (int u = 0; u < 4; ++u) {
      int i = tid + (g + u) * 128;
      int r = i >> 3, c4 = (i & 7) << 2;
      *(v4h*)&sK[r][c4] = cvt4(kr[u]);
    }
  }
  __syncthreads();

  const float scale = 0.17677669529663687f;  // 1/sqrt(32)
  {
    v16h a0 = frag_h(&sQ[(lane & 15)][0], lane >> 4);
    v16h a1 = frag_h(&sQ[16 + (lane & 15)][0], lane >> 4);
#pragma unroll
    for (int j = 0; j < 4; ++j) {
      int n0 = wv * 64 + j * 16;
      v16h bf = frag_h(&sK[n0 + (lane & 15)][0], lane >> 4);
      v8f s0 = {}; v8f s1 = {};
      s0 = wmma32(a0, bf, s0);
      s1 = wmma32(a1, bf, s1);
      int n = n0 + (lane & 15);
      int mb = ((lane >> 4) << 3);
#pragma unroll
      for (int r = 0; r < 8; ++r) {
        sS[mb + r][n]      = s0[r] * scale;
        sS[16 + mb + r][n] = s1[r] * scale;
      }
    }
  }
  __syncthreads();

  if (tid < 32) {  // row-wise softmax over 256 keys
    float mx = -1e30f;
    for (int k = 0; k < 256; ++k) mx = fmaxf(mx, sS[tid][k]);
    float sm = 0.f;
    for (int k = 0; k < 256; ++k) { float e = __expf(sS[tid][k] - mx); sS[tid][k] = e; sm += e; }
    float inv = 1.f / sm;
    for (int k = 0; k < 256; ++k) sS[tid][k] *= inv;
  }
  __syncthreads();

  // Stage V transposed (overwrites sK slot), batched in groups of 4
#pragma unroll
  for (int g = 0; g < 16; g += 4) {
    float4 vr[4];
#pragma unroll
    for (int u = 0; u < 4; ++u) {
      int i = tid + (g + u) * 128;
      int kk = i >> 3, d4 = (i & 7) << 2;
      vr[u] = *(const float4*)&qkv[(size_t)(b * S_ + kk) * 768 + 512 + hh * 32 + d4];
    }
#pragma unroll
    for (int u = 0; u < 4; ++u) {
      int i = tid + (g + u) * 128;
      int kk = i >> 3, d4 = (i & 7) << 2;
      sVt[d4 + 0][kk] = (_Float16)vr[u].x;
      sVt[d4 + 1][kk] = (_Float16)vr[u].y;
      sVt[d4 + 2][kk] = (_Float16)vr[u].z;
      sVt[d4 + 3][kk] = (_Float16)vr[u].w;
    }
  }
  __syncthreads();

  const int mi = wv & 1, nj = wv >> 1;  // 4 waves cover 32x32 output
  v8f o = {};
  for (int kt = 0; kt < 256; kt += 32) {
    v16h a  = frag_f(&sS[mi * 16 + (lane & 15)][kt], lane >> 4);
    v16h bf = frag_h(&sVt[nj * 16 + (lane & 15)][kt], lane >> 4);
    o = wmma32(a, bf, o);
  }
  int n  = hh * 32 + nj * 16 + (lane & 15);
  int mb = q0 + mi * 16 + ((lane >> 4) << 3);
#pragma unroll
  for (int r = 0; r < 8; ++r)
    out[(size_t)(b * S_ + mb + r) * H_ + n] = o[r];
}

// ---------------------------------------------------------------------------
// LayerNorm over H=256; one wave32 per row, 8 rows per 256-thread block.
// ---------------------------------------------------------------------------
__global__ __launch_bounds__(256) void ln_kernel(
    const float* __restrict__ src, const float* __restrict__ g,
    const float* __restrict__ bt, float* __restrict__ dst)
{
  int row = blockIdx.x * 8 + (threadIdx.x >> 5);
  int lane = threadIdx.x & 31;
  const float* x = src + (size_t)row * H_;
  float vals[8];
  float s = 0.f;
#pragma unroll
  for (int i = 0; i < 8; ++i) { vals[i] = x[lane + i * 32]; s += vals[i]; }
  for (int m = 16; m > 0; m >>= 1) s += __shfl_xor(s, m, 32);
  float mean = s * (1.f / 256.f);
  float v = 0.f;
#pragma unroll
  for (int i = 0; i < 8; ++i) { float d = vals[i] - mean; v += d * d; }
  for (int m = 16; m > 0; m >>= 1) v += __shfl_xor(v, m, 32);
  float inv = rsqrtf(v * (1.f / 256.f) + 1e-5f);
  float* y = dst + (size_t)row * H_;
#pragma unroll
  for (int i = 0; i < 8; ++i) {
    int c = lane + i * 32;
    y[c] = (vals[i] - mean) * inv * g[c] + bt[c];
  }
}

// Bin encoder: [32,2] -> relu -> [32,64] -> [32,64]; one block per batch row.
__global__ void bin_kernel(const float* __restrict__ bin,
                           const float* __restrict__ W1, const float* __restrict__ b1,
                           const float* __restrict__ W2, const float* __restrict__ b2,
                           float* __restrict__ bb)
{
  __shared__ float t1[64];
  int b = blockIdx.x, j = threadIdx.x;  // 64 threads
  float a = W1[j * 2 + 0] * bin[b * 2 + 0] + W1[j * 2 + 1] * bin[b * 2 + 1] + b1[j];
  t1[j] = fmaxf(a, 0.f);
  __syncthreads();
  float s = b2[j];
  for (int k = 0; k < 64; ++k) s += W2[j * 64 + k] * t1[k];
  bb[b * 64 + j] = s;
}

// Fill x[:, :, 128:256] with broadcast bin embedding + positional embedding.
__global__ void assemble_kernel(const float* __restrict__ bb,
                                const float* __restrict__ pos,
                                float* __restrict__ x)
{
  int row = blockIdx.x;               // 0..8191 = b*256+s
  int b = row >> 8, ss = row & 255;
  int j = threadIdx.x;                // 128 threads
  float v = (j < 64) ? bb[b * 64 + j] : pos[ss * 64 + (j - 64)];
  x[(size_t)row * H_ + 128 + j] = v;
}

// Teacher-forcing gather: xin[b,0]=0 ; xin[b,t]=enc[b, target[b,t-1]].
__global__ void gather_kernel(const float* __restrict__ enc,
                              const int* __restrict__ target,
                              float* __restrict__ xin)
{
  int row = blockIdx.x;  // b*256+t
  int b = row >> 8, t = row & 255;
  int j = threadIdx.x;   // 256 threads
  float v = 0.f;
  if (t > 0) {
    int tg = target[b * S_ + (t - 1)];
    v = enc[(size_t)(b * S_ + tg) * H_ + j];
  }
  xin[(size_t)row * H_ + j] = v;
}

// LSTM cell elementwise update (gate order i,f,g,o as in jnp.split).
__global__ void lstm_elem_kernel(const float* __restrict__ G,
                                 float* __restrict__ c, float* __restrict__ h)
{
  int idx = blockIdx.x * 256 + threadIdx.x;  // 0..8191
  int b = idx >> 8, j = idx & 255;
  const float* g = G + (size_t)b * 1024;
  float gi = g[j], gf = g[256 + j], gg = g[512 + j], go = g[768 + j];
  float si = 1.f / (1.f + __expf(-gi));
  float sf = 1.f / (1.f + __expf(-gf));
  float so = 1.f / (1.f + __expf(-go));
  float cc = sf * c[idx] + si * tanhf(gg);
  c[idx] = cc;
  h[idx] = so * tanhf(cc);
}

__global__ void zero_kernel(float* __restrict__ p, int n)
{
  int i = blockIdx.x * 256 + threadIdx.x;
  if (i < n) p[i] = 0.f;
}

// logits[b,t,i] = sum_j v[j] * tanh(hp[b,t,j] + ep[b,i,j])
// Block per (b, 32 t's, 32 i's); both tiles staged transposed in LDS (64KB).
__global__ __launch_bounds__(256) void logits_kernel(
    const float* __restrict__ hp, const float* __restrict__ ep,
    const float* __restrict__ v, float* __restrict__ out)
{
  __shared__ float sHT[256][32];
  __shared__ float sET[256][32];
  const int b = blockIdx.z, t0 = blockIdx.y * 32, i0 = blockIdx.x * 32;
  const int tid = threadIdx.x;
  // 32 rows x 64 float4 per tile = 2048 items, 256 threads -> 8 each;
  // batch loads in groups of 4 (8 outstanding b128 loads per group).
#pragma unroll
  for (int g = 0; g < 8; g += 4) {
    float4 hr[4], er[4];
#pragma unroll
    for (int u = 0; u < 4; ++u) {
      int i = tid + (g + u) * 256;
      int r = i >> 6, j = (i & 63) << 2;
      hr[u] = *(const float4*)&hp[(size_t)(b * S_ + t0 + r) * H_ + j];
      er[u] = *(const float4*)&ep[(size_t)(b * S_ + i0 + r) * H_ + j];
    }
#pragma unroll
    for (int u = 0; u < 4; ++u) {
      int i = tid + (g + u) * 256;
      int r = i >> 6, j = (i & 63) << 2;
      sHT[j][r] = hr[u].x; sHT[j + 1][r] = hr[u].y;
      sHT[j + 2][r] = hr[u].z; sHT[j + 3][r] = hr[u].w;
      sET[j][r] = er[u].x; sET[j + 1][r] = er[u].y;
      sET[j + 2][r] = er[u].z; sET[j + 3][r] = er[u].w;
    }
  }
  __syncthreads();
  const int tl = tid >> 3;    // local t (0..31)
  const int is = tid & 7;     // i sub-lane
  float acc[4] = {0.f, 0.f, 0.f, 0.f};
  for (int j = 0; j < 256; ++j) {
    float vh = sHT[j][tl];
    float vj = v[j];
#pragma unroll
    for (int q = 0; q < 4; ++q)
      acc[q] += vj * tanhf(vh + sET[j][is + q * 8]);
  }
#pragma unroll
  for (int q = 0; q < 4; ++q)
    out[(size_t)(b * S_ + t0 + tl) * S_ + i0 + is + q * 8] = acc[q];
}

// ---------------------------------------------------------------------------
// Host orchestration
// ---------------------------------------------------------------------------
extern "C" void kernel_launch(void* const* d_in, const int* in_sizes, int n_in,
                              void* d_out, int out_size, void* d_ws, size_t ws_size,
                              hipStream_t stream) {
  (void)in_sizes; (void)n_in; (void)out_size; (void)ws_size;
  const float* parts    = (const float*)d_in[0];
  const float* bin_info = (const float*)d_in[1];
  const int*   target   = (const int*)d_in[2];
  const float* pe_W1 = (const float*)d_in[3];
  const float* pe_b1 = (const float*)d_in[4];
  const float* pe_W2 = (const float*)d_in[5];
  const float* pe_b2 = (const float*)d_in[6];
  const float* be_W1 = (const float*)d_in[7];
  const float* be_b1 = (const float*)d_in[8];
  const float* be_W2 = (const float*)d_in[9];
  const float* be_b2 = (const float*)d_in[10];
  const float* pos_emb = (const float*)d_in[11];
  const float* tr_Wqkv = (const float*)d_in[12];
  const float* tr_bqkv = (const float*)d_in[13];
  const float* tr_Wo   = (const float*)d_in[14];
  const float* tr_bo   = (const float*)d_in[15];
  const float* tr_ln1g = (const float*)d_in[16];
  const float* tr_ln1b = (const float*)d_in[17];
  const float* tr_ffW1 = (const float*)d_in[18];
  const float* tr_ffb1 = (const float*)d_in[19];
  const float* tr_ffW2 = (const float*)d_in[20];
  const float* tr_ffb2 = (const float*)d_in[21];
  const float* tr_ln2g = (const float*)d_in[22];
  const float* tr_ln2b = (const float*)d_in[23];
  const float* lstm_Wih = (const float*)d_in[24];
  const float* lstm_Whh = (const float*)d_in[25];
  const float* lstm_bih = (const float*)d_in[26];
  const float* lstm_bhh = (const float*)d_in[27];
  const float* ptr_W = (const float*)d_in[28];
  const float* ptr_b = (const float*)d_in[29];
  const float* ptr_v = (const float*)d_in[30];

  // Workspace carve-up (all fp32)
  char* ws = (char*)d_ws;
  float* x     = (float*)ws; ws += (size_t)M_ * H_   * 4;  // activations
  float* y     = (float*)ws; ws += (size_t)M_ * H_   * 4;  // pre-LN buffer
  float* big   = (float*)ws; ws += (size_t)M_ * 1024 * 4;  // qkv / ff / gates_ih
  float* cbuf  = (float*)ws; ws += (size_t)M_ * H_   * 4;  // attn_out / xin
  float* eproj = (float*)ws; ws += (size_t)M_ * H_   * 4;
  float* hp    = (float*)ws; ws += (size_t)M_ * H_   * 4;
  float* bb    = (float*)ws; ws += (size_t)B_ * 64   * 4;
  float* hbuf  = (float*)ws; ws += (size_t)B_ * H_   * 4;
  float* ccell = (float*)ws; ws += (size_t)B_ * H_   * 4;
  float* G     = (float*)ws; ws += (size_t)B_ * 1024 * 4;

  const dim3 blk128(128), blk256(256);

  // --- part encoder MLP: relu(parts @ W1^T + b1) @ W2^T + b2 -> x[:, :128]
  gemm_wmma_kernel<true><<<dim3(2, M_ / 64), blk128, 0, stream>>>(
      parts, DIN_, pe_W1, DIN_, pe_b1, nullptr, 0, big, 128, DIN_, 1);
  gemm_wmma_kernel<false><<<dim3(2, M_ / 64), blk128, 0, stream>>>(
      big, 128, pe_W2, 128, pe_b2, nullptr, 0, x, H_, 128, 0);
  // --- bin encoder + broadcast + positional -> x[:, 128:256]
  bin_kernel<<<B_, 64, 0, stream>>>(bin_info, be_W1, be_b1, be_W2, be_b2, bb);
  assemble_kernel<<<M_, 128, 0, stream>>>(bb, pos_emb, x);

  // --- transformer encoder layers
  for (int l = 0; l < NL_; ++l) {
    const float* Wqkv = tr_Wqkv + (size_t)l * 768 * H_;
    const float* bqkv = tr_bqkv + (size_t)l * 768;
    const float* Wo   = tr_Wo   + (size_t)l * H_ * H_;
    const float* bo   = tr_bo   + (size_t)l * H_;
    const float* W1   = tr_ffW1 + (size_t)l * 512 * H_;
    const float* b1   = tr_ffb1 + (size_t)l * 512;
    const float* W2   = tr_ffW2 + (size_t)l * H_ * 512;
    const float* b2   = tr_ffb2 + (size_t)l * H_;

    gemm_wmma_kernel<false><<<dim3(12, M_ / 64), blk128, 0, stream>>>(
        x, H_, Wqkv, H_, bqkv, nullptr, 0, big, 768, H_, 0);
    attn_kernel<<<dim3(S_ / 32, NH_, B_), blk128, 0, stream>>>(big, cbuf);
    gemm_wmma_kernel<false><<<dim3(4, M_ / 64), blk128, 0, stream>>>(
        cbuf, H_, Wo, H_, bo, x, H_, y, H_, H_, 0);
    ln_kernel<<<M_ / 8, blk256, 0, stream>>>(y, tr_ln1g + l * H_, tr_ln1b + l * H_, x);
    gemm_wmma_kernel<false><<<dim3(8, M_ / 64), blk128, 0, stream>>>(
        x, H_, W1, H_, b1, nullptr, 0, big, 512, H_, 1);
    gemm_wmma_kernel<false><<<dim3(4, M_ / 64), blk128, 0, stream>>>(
        big, 512, W2, 512, b2, x, H_, y, H_, 512, 0);
    ln_kernel<<<M_ / 8, blk256, 0, stream>>>(y, tr_ln2g + l * H_, tr_ln2b + l * H_, x);
  }

  // --- pointer projections of the encoder output
  gemm_wmma_kernel<false><<<dim3(4, M_ / 64), blk128, 0, stream>>>(
      x, H_, ptr_W, H_, ptr_b, nullptr, 0, eproj, H_, H_, 0);

  // --- hoist teacher-forcing inputs and their Wih projection (one big GEMM)
  gather_kernel<<<M_, blk256, 0, stream>>>(x, target, cbuf);
  gemm_wmma_kernel<false><<<dim3(16, M_ / 64), blk128, 0, stream>>>(
      cbuf, H_, lstm_Wih, H_, lstm_bih, nullptr, 0, big, 1024, H_, 0);

  // --- sequential LSTM pointer decode
  zero_kernel<<<(2 * B_ * H_ + 255) / 256, blk256, 0, stream>>>(hbuf, 2 * B_ * H_);
  for (int t = 0; t < S_; ++t) {
    gemm_small_kernel<<<16, blk128, 0, stream>>>(
        hbuf, H_, lstm_Whh, H_, lstm_bhh, big + (size_t)t * 1024, S_ * 1024,
        G, 1024, H_);
    lstm_elem_kernel<<<(B_ * H_) / 256, blk256, 0, stream>>>(G, ccell, hbuf);
    gemm_small_kernel<<<4, blk128, 0, stream>>>(
        hbuf, H_, ptr_W, H_, ptr_b, nullptr, 0, hp + (size_t)t * H_, S_ * H_, H_);
  }

  // --- final pointer logits [B,S,S]
  logits_kernel<<<dim3(S_ / 32, S_ / 32, B_), blk256, 0, stream>>>(
      hp, eproj, ptr_v, (float*)d_out);
}
